// GCNGlobal_33552284516576
// MI455X (gfx1250) — matrix operand order repaired and checked by reading, not compiled, not run on hardware
//
#include <hip/hip_runtime.h>
#include <hip/hip_bf16.h>
#include <cstdint>

typedef _Float16 v16h __attribute__((ext_vector_type(16)));
typedef _Float16 v8h  __attribute__((ext_vector_type(8)));
typedef float    v8f  __attribute__((ext_vector_type(8)));

#define B_    16
#define N_    2048
#define F_    64
#define HD_   256
#define K_    128
#define G_    18
#define FUSE_ 274
#define LN_EPS 1e-5f

// ---- workspace layout (bytes) ----
#define XT_OFF   ((size_t)0)
#define W1T_OFF  (XT_OFF  + (size_t)B_*F_*N_*2)    // Xt  f16 (B,F,N)   4 MB
#define W2T_OFF  (W1T_OFF + (size_t)HD_*F_*2)      // W1t f16 (HD,F)    32 KB
#define H1T_OFF  (W2T_OFF + (size_t)HD_*HD_*2)     // W2t f16 (HD,HD)   128 KB
#define GP_OFF   (H1T_OFF + (size_t)B_*HD_*N_*2)   // H1t f16 (B,HD,N)  16 MB
// gpart f32: 512*HD_ floats                        // 512 KB

__device__ __forceinline__ float lane_xor_add(float v, int mask, int lane) {
  // xor-shuffle add via ds_bpermute (wave32: index bits [6:2] used)
  int o = __builtin_amdgcn_ds_bpermute(((lane ^ mask) << 2), __float_as_int(v));
  return v + __int_as_float(o);
}

__device__ __forceinline__ v8f wmma_f16(v16h a, v16h b, v8f c) {
  return __builtin_amdgcn_wmma_f32_16x16x32_f16(false, a, false, b, (short)0, c,
                                                false, false);
}

__device__ __forceinline__ v16h cvt16(float4 a0, float4 a1, float4 a2, float4 a3) {
  v16h a;
  a[0]=(_Float16)a0.x; a[1]=(_Float16)a0.y; a[2]=(_Float16)a0.z; a[3]=(_Float16)a0.w;
  a[4]=(_Float16)a1.x; a[5]=(_Float16)a1.y; a[6]=(_Float16)a1.z; a[7]=(_Float16)a1.w;
  a[8]=(_Float16)a2.x; a[9]=(_Float16)a2.y; a[10]=(_Float16)a2.z; a[11]=(_Float16)a2.w;
  a[12]=(_Float16)a3.x; a[13]=(_Float16)a3.y; a[14]=(_Float16)a3.z; a[15]=(_Float16)a3.w;
  return a;
}

// ---------------- prep: transpose + f16 convert X, W1, W2 ----------------
__global__ void __launch_bounds__(256)
prep_kernel(const float* __restrict__ X, const float* __restrict__ W1,
            const float* __restrict__ W2, _Float16* __restrict__ Xt,
            _Float16* __restrict__ W1t, _Float16* __restrict__ W2t) {
  size_t gtid = (size_t)blockIdx.x * 256 + threadIdx.x;
  size_t stride = (size_t)gridDim.x * 256;
  for (size_t i = gtid; i < (size_t)B_ * F_ * N_; i += stride) {
    int n = (int)(i % N_);
    int f = (int)((i / N_) % F_);
    int b = (int)(i / ((size_t)F_ * N_));
    Xt[i] = (_Float16)X[((size_t)b * N_ + n) * F_ + f];
  }
  for (size_t i = gtid; i < (size_t)HD_ * F_; i += stride) {
    int f = (int)(i % F_);
    int h = (int)(i / F_);
    W1t[i] = (_Float16)W1[(size_t)f * HD_ + h];
  }
  for (size_t i = gtid; i < (size_t)HD_ * HD_; i += stride) {
    int h1 = (int)(i % HD_);
    int h2 = (int)(i / HD_);
    W2t[i] = (_Float16)W2[(size_t)h1 * HD_ + h2];
  }
}

// ---------------- layer 1: H1 = relu(LN((A@X)@W1 + b1)) -> H1t f16 ----------------
__global__ void __launch_bounds__(128)
gcn_layer1(const float* __restrict__ A, const _Float16* __restrict__ Xt,
           const _Float16* __restrict__ W1t, const float* __restrict__ b1,
           const float* __restrict__ g1, const float* __restrict__ beta1,
           _Float16* __restrict__ H1t) {
  __shared__ __align__(16) _Float16 yld[4][16][F_ + 8];   // Y1 restripe buffer
  const int tid  = threadIdx.x;
  const int wave = tid >> 5, lane = tid & 31;
  const int r = lane & 15, kh = lane >> 4;
  const int b = blockIdx.x >> 5;            // 32 row-blocks per batch
  const int rowBlk = blockIdx.x & 31;
  const int rowBase = rowBlk * 64 + wave * 16;

  const float* Arow = A + ((size_t)b * N_ + (rowBase + r)) * (size_t)N_;

  // Phase 1: Y1(16x64) = A[rows,:]@X  (K loop over N)
  v8f accY[4] = {};
  #pragma unroll 2
  for (int k = 0; k < N_; k += 32) {
    const float* ap = Arow + k + kh * 16;  // A-operand: K contiguous (k = 16*kh + s)
    __builtin_prefetch(ap + 64, 0, 1);
    float4 a0 = ((const float4*)ap)[0];
    float4 a1 = ((const float4*)ap)[1];
    float4 a2 = ((const float4*)(ap + 8))[0];
    float4 a3 = ((const float4*)(ap + 8))[1];
    v16h a = cvt16(a0, a1, a2, a3);
    #pragma unroll
    for (int t = 0; t < 4; ++t) {
      const _Float16* bp = Xt + ((size_t)b * F_ + t * 16 + r) * N_ + k + kh * 16;
      v16h bv = *(const v16h*)bp;          // B-operand: col=lane&15, K contiguous
      accY[t] = wmma_f16(a, bv, accY[t]);
    }
  }

  // Phase 2: restripe Y1 (C-layout -> row-major f16 in LDS)
  #pragma unroll
  for (int t = 0; t < 4; ++t)
    #pragma unroll
    for (int j = 0; j < 8; ++j)
      yld[wave][j + 8 * kh][t * 16 + r] = (_Float16)accY[t][j];
  __syncthreads();

  // Phase 3: G1(16x256) = Y1 @ W1
  v8f accG[16];
  #pragma unroll
  for (int tt = 0; tt < 16; ++tt) accG[tt] = (v8f){};
  #pragma unroll
  for (int kk = 0; kk < F_; kk += 32) {
    int f0 = kk + kh * 16;
    v8h alo = *(const v8h*)&yld[wave][r][f0];
    v8h ahi = *(const v8h*)&yld[wave][r][f0 + 8];
    v16h a;
    #pragma unroll
    for (int i = 0; i < 8; ++i) { a[i] = alo[i]; a[8 + i] = ahi[i]; }
    #pragma unroll
    for (int tt = 0; tt < 16; ++tt) {
      const _Float16* wp = W1t + (size_t)(tt * 16 + r) * F_ + kk + kh * 16;
      v16h bw = *(const v16h*)wp;
      accG[tt] = wmma_f16(a, bw, accG[tt]);
    }
  }

  // Phase 4: +bias, LayerNorm over 256, ReLU, store H1t (transposed f16)
  float mu[8], rs[8];
  #pragma unroll
  for (int j = 0; j < 8; ++j) {
    float s = 0.f, ss = 0.f;
    #pragma unroll
    for (int tt = 0; tt < 16; ++tt) {
      float v = accG[tt][j] + b1[tt * 16 + r];
      accG[tt][j] = v;
      s += v; ss += v * v;
    }
    s  = lane_xor_add(s, 1, lane);  s  = lane_xor_add(s, 2, lane);
    s  = lane_xor_add(s, 4, lane);  s  = lane_xor_add(s, 8, lane);
    ss = lane_xor_add(ss, 1, lane); ss = lane_xor_add(ss, 2, lane);
    ss = lane_xor_add(ss, 4, lane); ss = lane_xor_add(ss, 8, lane);
    float m = s * (1.0f / HD_);
    mu[j] = m;
    rs[j] = rsqrtf(fmaxf(ss * (1.0f / HD_) - m * m, 0.f) + LN_EPS);
  }
  #pragma unroll
  for (int tt = 0; tt < 16; ++tt) {
    int col = tt * 16 + r;
    float gg = g1[col], bb = beta1[col];
    v8h o;
    #pragma unroll
    for (int j = 0; j < 8; ++j) {
      float x = (accG[tt][j] - mu[j]) * rs[j] * gg + bb;
      o[j] = (_Float16)fmaxf(x, 0.f);
    }
    *(v8h*)(H1t + ((size_t)b * HD_ + col) * N_ + rowBase + kh * 8) = o;
  }
}

// ---------------- layer 2: H2 = relu(LN((A@H1)@W2 + b2)); emit column partial sums ----------------
__global__ void __launch_bounds__(128)
gcn_layer2(const float* __restrict__ A, const _Float16* __restrict__ H1t,
           const _Float16* __restrict__ W2t, const float* __restrict__ b2,
           const float* __restrict__ g2, const float* __restrict__ beta2,
           float* __restrict__ gpart) {
  __shared__ __align__(16) _Float16 yld[4][16][HD_ + 8];  // Y2 restripe buffer
  __shared__ float red[4][HD_];
  const int tid  = threadIdx.x;
  const int wave = tid >> 5, lane = tid & 31;
  const int r = lane & 15, kh = lane >> 4;
  const int b = blockIdx.x >> 5;
  const int rowBlk = blockIdx.x & 31;
  const int rowBase = rowBlk * 64 + wave * 16;

  const float* Arow = A + ((size_t)b * N_ + (rowBase + r)) * (size_t)N_;

  // Phase 1: Y2(16x256) = A[rows,:] @ H1
  v8f accY[16];
  #pragma unroll
  for (int t = 0; t < 16; ++t) accY[t] = (v8f){};
  #pragma unroll 1
  for (int k = 0; k < N_; k += 32) {
    const float* ap = Arow + k + kh * 16;
    __builtin_prefetch(ap + 64, 0, 1);
    float4 a0 = ((const float4*)ap)[0];
    float4 a1 = ((const float4*)ap)[1];
    float4 a2 = ((const float4*)(ap + 8))[0];
    float4 a3 = ((const float4*)(ap + 8))[1];
    v16h a = cvt16(a0, a1, a2, a3);
    #pragma unroll
    for (int t = 0; t < 16; ++t) {
      const _Float16* bp = H1t + ((size_t)b * HD_ + t * 16 + r) * N_ + k + kh * 16;
      v16h bv = *(const v16h*)bp;
      accY[t] = wmma_f16(a, bv, accY[t]);
    }
  }

  // Phase 2: restripe Y2 into LDS f16 (free the 128 acc VGPRs)
  #pragma unroll
  for (int t = 0; t < 16; ++t)
    #pragma unroll
    for (int j = 0; j < 8; ++j)
      yld[wave][j + 8 * kh][t * 16 + r] = (_Float16)accY[t][j];
  __syncthreads();

  // Phase 3: G2(16x256) = Y2 @ W2
  v8f accG[16];
  #pragma unroll
  for (int tt = 0; tt < 16; ++tt) accG[tt] = (v8f){};
  #pragma unroll 1
  for (int kk = 0; kk < HD_; kk += 32) {
    int f0 = kk + kh * 16;
    v8h alo = *(const v8h*)&yld[wave][r][f0];
    v8h ahi = *(const v8h*)&yld[wave][r][f0 + 8];
    v16h a;
    #pragma unroll
    for (int i = 0; i < 8; ++i) { a[i] = alo[i]; a[8 + i] = ahi[i]; }
    #pragma unroll
    for (int tt = 0; tt < 16; ++tt) {
      const _Float16* wp = W2t + (size_t)(tt * 16 + r) * HD_ + kk + kh * 16;
      v16h bw = *(const v16h*)wp;
      accG[tt] = wmma_f16(a, bw, accG[tt]);
    }
  }

  // Phase 4: +bias, LN, ReLU; accumulate mean-readout partial sums (deterministic)
  float mu[8], rs[8];
  #pragma unroll
  for (int j = 0; j < 8; ++j) {
    float s = 0.f, ss = 0.f;
    #pragma unroll
    for (int tt = 0; tt < 16; ++tt) {
      float v = accG[tt][j] + b2[tt * 16 + r];
      accG[tt][j] = v;
      s += v; ss += v * v;
    }
    s  = lane_xor_add(s, 1, lane);  s  = lane_xor_add(s, 2, lane);
    s  = lane_xor_add(s, 4, lane);  s  = lane_xor_add(s, 8, lane);
    ss = lane_xor_add(ss, 1, lane); ss = lane_xor_add(ss, 2, lane);
    ss = lane_xor_add(ss, 4, lane); ss = lane_xor_add(ss, 8, lane);
    float m = s * (1.0f / HD_);
    mu[j] = m;
    rs[j] = rsqrtf(fmaxf(ss * (1.0f / HD_) - m * m, 0.f) + LN_EPS);
  }
  #pragma unroll
  for (int tt = 0; tt < 16; ++tt) {
    int col = tt * 16 + r;
    float gg = g2[col], bb = beta2[col];
    float cs = 0.f;
    #pragma unroll
    for (int j = 0; j < 8; ++j) {
      float x = (accG[tt][j] - mu[j]) * rs[j] * gg + bb;
      cs += fmaxf(x, 0.f);
    }
    cs = lane_xor_add(cs, 16, lane);     // combine halves -> sum over 16 rows
    if (kh == 0) red[wave][col] = cs;
  }
  __syncthreads();
  for (int col = tid; col < HD_; col += 128) {
    gpart[(size_t)blockIdx.x * HD_ + col] =
        red[0][col] + red[1][col] + red[2][col] + red[3][col];
  }
}

// ---------------- head: mean-readout reduce, concat, two small GEMVs ----------------
__global__ void __launch_bounds__(256)
head_kernel(const float* __restrict__ gpart, const float* __restrict__ gvec,
            const float* __restrict__ Ws, const float* __restrict__ bs,
            const float* __restrict__ Wa, const float* __restrict__ ba,
            float* __restrict__ out) {
  __shared__ float fused[B_][FUSE_];
  int tid = threadIdx.x;
  for (int i = tid; i < B_ * HD_; i += 256) {
    int b = i >> 8, col = i & 255;
    float s = 0.f;
    for (int rb = 0; rb < 32; ++rb)
      s += gpart[(size_t)(b * 32 + rb) * HD_ + col];
    fused[b][col] = s * (1.0f / N_);
  }
  for (int i = tid; i < B_ * G_; i += 256) {
    int b = i / G_, g = i % G_;
    fused[b][HD_ + g] = gvec[i];
  }
  __syncthreads();
  if (tid < B_) {
    float s = bs[0];
    for (int i = 0; i < FUSE_; ++i) s += fused[tid][i] * Ws[i];
    out[tid] = s;
  }
  for (int idx = tid; idx < B_ * K_; idx += 256) {
    int b = idx >> 7, kk = idx & 127;
    float s = ba[kk];
    for (int i = 0; i < FUSE_; ++i) s += fused[b][i] * Wa[i * K_ + kk];
    out[B_ + idx] = s;
  }
}

extern "C" void kernel_launch(void* const* d_in, const int* in_sizes, int n_in,
                              void* d_out, int out_size, void* d_ws, size_t ws_size,
                              hipStream_t stream) {
  (void)in_sizes; (void)n_in; (void)out_size; (void)ws_size;
  const float* A    = (const float*)d_in[0];
  const float* X    = (const float*)d_in[1];
  const float* gvec = (const float*)d_in[2];
  const float* W1   = (const float*)d_in[3];
  const float* b1   = (const float*)d_in[4];
  const float* g1   = (const float*)d_in[5];
  const float* be1  = (const float*)d_in[6];
  const float* W2   = (const float*)d_in[7];
  const float* b2   = (const float*)d_in[8];
  const float* g2   = (const float*)d_in[9];
  const float* be2  = (const float*)d_in[10];
  const float* Ws   = (const float*)d_in[11];
  const float* bs   = (const float*)d_in[12];
  const float* Wa   = (const float*)d_in[13];
  const float* ba   = (const float*)d_in[14];
  float* out = (float*)d_out;

  char* ws = (char*)d_ws;
  _Float16* Xt  = (_Float16*)(ws + XT_OFF);
  _Float16* W1t = (_Float16*)(ws + W1T_OFF);
  _Float16* W2t = (_Float16*)(ws + W2T_OFF);
  _Float16* H1t = (_Float16*)(ws + H1T_OFF);
  float*    gpart = (float*)(ws + GP_OFF);

  prep_kernel<<<1024, 256, 0, stream>>>(X, W1, W2, Xt, W1t, W2t);
  gcn_layer1<<<B_ * (N_ / 64), 128, 0, stream>>>(A, Xt, W1t, b1, g1, be1, H1t);
  gcn_layer2<<<B_ * (N_ / 64), 128, 0, stream>>>(A, H1t, W2t, b2, g2, be2, gpart);
  head_kernel<<<1, 256, 0, stream>>>(gpart, gvec, Ws, bs, Wa, ba, out);
}
